// CQAttention_687194767975
// MI455X (gfx1250) — matrix operand order repaired and checked by reading, not compiled, not run on hardware
//
#include <hip/hip_runtime.h>
#include <hip/hip_bf16.h>

typedef _Float16 half_t;
typedef __attribute__((ext_vector_type(16))) _Float16 v16h;
typedef __attribute__((ext_vector_type(8)))  _Float16 v8h;
typedef __attribute__((ext_vector_type(4)))  _Float16 v4h;
typedef __attribute__((ext_vector_type(8)))  float    v8f;

#define B_DIM 16
#define LC    1024
#define LQ    256
#define DDIM  512
#define KB    32
#define LDT   40   // LDS row stride in halfs for K=32 tiles (80B, 16B-aligned)

// ---------------------------------------------------------------------------
// WMMA helpers (gfx1250 wave32, v_wmma_f32_16x16x32_f16)
// ---------------------------------------------------------------------------
__device__ inline v8f wmma16(v16h a, v16h b, v8f c) {
    return __builtin_amdgcn_wmma_f32_16x16x32_f16(
        /*neg_a=*/false, a, /*neg_b=*/false, b,
        /*c_mod=*/(short)0, c, /*reuse_a=*/false, /*reuse_b=*/false);
}

// A fragment: 16x32 f16, tile stored row-major [M][K] with stride LDT.
// lane: m = lane&15, kb = (lane>=16)?8:0; frag[0..7]=K kb..kb+7, frag[8..15]=K kb+16..kb+23
__device__ inline v16h frag_a(const half_t* tile, int lane) {
    int m  = lane & 15;
    int kb = (lane >> 4) << 3;
    const half_t* row = tile + m * LDT + kb;
    v8h lo = *(const v8h*)(row);
    v8h hi = *(const v8h*)(row + 16);
    v16h a;
#pragma unroll
    for (int i = 0; i < 8; ++i) { a[i] = lo[i]; a[8 + i] = hi[i]; }
    return a;
}

// B fragment: 32x16 f16, tile stored transposed [N][K] with stride LDT.
// lane: n = lane&15, kb = (lane>=16)?8:0; frag[0..7]=K kb..kb+7, frag[8..15]=K kb+16..kb+23
__device__ inline v16h frag_b(const half_t* tile, int lane) {
    int n  = lane & 15;
    int kb = (lane >> 4) << 3;
    const half_t* row = tile + n * LDT + kb;
    v8h lo = *(const v8h*)(row);
    v8h hi = *(const v8h*)(row + 16);
    v16h b;
#pragma unroll
    for (int i = 0; i < 8; ++i) { b[i] = lo[i]; b[8 + i] = hi[i]; }
    return b;
}

// ---------------------------------------------------------------------------
// K0: cw[b,i] = dot(C[b,i,:], w4C) + bias ; qw[b,j] = dot(Q[b,j,:], w4Q)
// ---------------------------------------------------------------------------
__global__ __launch_bounds__(256)
void k_rowdots(const float* __restrict__ C, const float* __restrict__ Q,
               const float* __restrict__ w4C, const float* __restrict__ w4Q,
               const float* __restrict__ bias,
               float* __restrict__ cw, float* __restrict__ qw) {
    __shared__ float red[256];
    int idx = blockIdx.x;
    const float* row; const float* w; float* out; float add;
    if (idx < B_DIM * LC) {
        row = C + (size_t)idx * DDIM; w = w4C; out = cw + idx; add = bias[0];
    } else {
        idx -= B_DIM * LC;
        row = Q + (size_t)idx * DDIM; w = w4Q; out = qw + idx; add = 0.f;
    }
    int t = threadIdx.x;
    red[t] = row[t] * w[t] + row[t + 256] * w[t + 256];
    __syncthreads();
    for (int off = 128; off > 0; off >>= 1) {
        if (t < off) red[t] += red[t + off];
        __syncthreads();
    }
    if (t == 0) *out = red[0] + add;
}

// ---------------------------------------------------------------------------
// K1: S[b] = (C[b]*w4mlu) @ Q[b]^T + cw + qw   (NT GEMM, f16 WMMA, f32 out)
// grid (LQ/128, LC/128, B), block 256 (8 waves, 4x2), wave tile 32x64
// ---------------------------------------------------------------------------
__global__ __launch_bounds__(256)
void k_gemm_S(const float* __restrict__ C, const float* __restrict__ Q,
              const float* __restrict__ w4mlu, const float* __restrict__ cw,
              const float* __restrict__ qw, float* __restrict__ S) {
    __shared__ half_t aT[128 * LDT];
    __shared__ half_t bT[128 * LDT];
    const int b  = blockIdx.z;
    const int m0 = blockIdx.y * 128;
    const int n0 = blockIdx.x * 128;
    const int t    = threadIdx.x;
    const int lane = t & 31;
    const int wave = t >> 5;
    const int wm = wave >> 1;   // 0..3
    const int wn = wave & 1;    // 0..1

    const float* Cb = C + (size_t)b * LC * DDIM;
    const float* Qb = Q + (size_t)b * LQ * DDIM;

    v8f acc[2][4];
    const v8f z = {0.f,0.f,0.f,0.f,0.f,0.f,0.f,0.f};
#pragma unroll
    for (int i = 0; i < 2; ++i)
#pragma unroll
        for (int j = 0; j < 4; ++j) acc[i][j] = z;

    const int r  = t >> 3;          // 0..31
    const int ks = (t & 7) * 4;     // 0..28

    for (int k0 = 0; k0 < DDIM; k0 += KB) {
        __syncthreads();
        const float4 wv = *(const float4*)(w4mlu + k0 + ks);
#pragma unroll
        for (int i = 0; i < 4; ++i) {
            int rr = r + 32 * i;
            float4 v = *(const float4*)(Cb + (size_t)(m0 + rr) * DDIM + k0 + ks);
            half_t* d = aT + rr * LDT + ks;
            d[0] = (half_t)(v.x * wv.x); d[1] = (half_t)(v.y * wv.y);
            d[2] = (half_t)(v.z * wv.z); d[3] = (half_t)(v.w * wv.w);
            float4 q = *(const float4*)(Qb + (size_t)(n0 + rr) * DDIM + k0 + ks);
            half_t* e = bT + rr * LDT + ks;
            e[0] = (half_t)q.x; e[1] = (half_t)q.y; e[2] = (half_t)q.z; e[3] = (half_t)q.w;
        }
        __syncthreads();
        v16h af0 = frag_a(aT + (wm * 32 + 0)  * LDT, lane);
        v16h af1 = frag_a(aT + (wm * 32 + 16) * LDT, lane);
#pragma unroll
        for (int jn = 0; jn < 4; ++jn) {
            v16h bf = frag_b(bT + (wn * 64 + jn * 16) * LDT, lane);
            acc[0][jn] = wmma16(af0, bf, acc[0][jn]);
            acc[1][jn] = wmma16(af1, bf, acc[1][jn]);
        }
    }

    const int n_l = lane & 15;
    const int m_l = (lane >> 4) * 8;
    float* Sb = S + (size_t)b * LC * LQ;
#pragma unroll
    for (int im = 0; im < 2; ++im)
#pragma unroll
        for (int jn = 0; jn < 4; ++jn) {
            int nn = n0 + wn * 64 + jn * 16 + n_l;
            float qv = qw[b * LQ + nn];
#pragma unroll
            for (int v = 0; v < 8; ++v) {
                int mm = m0 + wm * 32 + im * 16 + m_l + v;
                Sb[(size_t)mm * LQ + nn] = acc[im][jn][v] + cw[b * LC + mm] + qv;
            }
        }
}

// ---------------------------------------------------------------------------
// K2a: row softmax over Lq with Qmask -> S1 (f16). grid B*LC, block 256
// ---------------------------------------------------------------------------
__global__ __launch_bounds__(256)
void k_softmax_rows(const float* __restrict__ S, const int* __restrict__ Qmask,
                    half_t* __restrict__ S1) {
    __shared__ float red[256];
    int row = blockIdx.x;           // b*LC + i
    int b   = row >> 10;            // LC = 1024
    int t   = threadIdx.x;          // j in [0,256)
    float qm = (float)Qmask[b * LQ + t];
    float s  = S[(size_t)row * LQ + t];
    s = fminf(fmaxf(s, -15.f), 15.f) * qm;
    red[t] = s; __syncthreads();
    for (int off = 128; off > 0; off >>= 1) {
        if (t < off) red[t] = fmaxf(red[t], red[t + off]);
        __syncthreads();
    }
    float mx = red[0]; __syncthreads();
    float e = __expf(s - mx) * qm;
    red[t] = e; __syncthreads();
    for (int off = 128; off > 0; off >>= 1) {
        if (t < off) red[t] += red[t + off];
        __syncthreads();
    }
    float denom = red[0] + 1e-6f;
    S1[(size_t)row * LQ + t] = (half_t)(e / denom);
}

// ---------------------------------------------------------------------------
// K2b: column softmax over Lc with Cmask -> S2^T (f16, [B,LQ,LC]).
// grid (LQ/16, B), block (16,16)
// ---------------------------------------------------------------------------
__global__ __launch_bounds__(256)
void k_softmax_cols(const float* __restrict__ S, const int* __restrict__ Cmask,
                    half_t* __restrict__ S2t) {
    __shared__ float red[16][17];
    const int b  = blockIdx.y;
    const int j  = blockIdx.x * 16 + threadIdx.x;
    const int ty = threadIdx.y;
    const int tx = threadIdx.x;
    const float* Sb = S + (size_t)b * LC * LQ;

    float x[64];
    unsigned long long bits = 0;
    float mx = -1e30f;
#pragma unroll
    for (int r = 0; r < 64; ++r) {
        int i = ty + r * 16;
        float cm = (float)Cmask[b * LC + i];
        float s  = Sb[(size_t)i * LQ + j];
        s = fminf(fmaxf(s, -15.f), 15.f) * cm;
        x[r] = s;
        if (cm != 0.f) bits |= (1ull << r);
        mx = fmaxf(mx, s);
    }
    red[ty][tx] = mx; __syncthreads();
    for (int off = 8; off > 0; off >>= 1) {
        if (ty < off) red[ty][tx] = fmaxf(red[ty][tx], red[ty + off][tx]);
        __syncthreads();
    }
    mx = red[0][tx]; __syncthreads();

    float sum = 0.f;
#pragma unroll
    for (int r = 0; r < 64; ++r) {
        float cm = (float)((bits >> r) & 1ull);
        float e = __expf(x[r] - mx) * cm;
        x[r] = e;
        sum += e;
    }
    red[ty][tx] = sum; __syncthreads();
    for (int off = 8; off > 0; off >>= 1) {
        if (ty < off) red[ty][tx] += red[ty + off][tx];
        __syncthreads();
    }
    float denom = red[0][tx] + 1e-6f;
    half_t* dst = S2t + ((size_t)b * LQ + j) * LC;
#pragma unroll
    for (int r = 0; r < 64; ++r) dst[ty + r * 16] = (half_t)(x[r] / denom);
}

// ---------------------------------------------------------------------------
// K3: T[b] = S2t[b] @ C[b]   ([LQ,LC] x [LC,D], NN GEMM) -> T16 (f16)
// grid (D/128, LQ/128, B), block 256
// ---------------------------------------------------------------------------
__global__ __launch_bounds__(256)
void k_gemm_T(const half_t* __restrict__ S2t, const float* __restrict__ C,
              half_t* __restrict__ T16) {
    __shared__ half_t aT[128 * LDT];
    __shared__ half_t bT[128 * LDT];
    const int b  = blockIdx.z;
    const int m0 = blockIdx.y * 128;   // over LQ
    const int n0 = blockIdx.x * 128;   // over D
    const int t = threadIdx.x;
    const int lane = t & 31;
    const int wave = t >> 5;
    const int wm = wave >> 1, wn = wave & 1;

    const half_t* Ab = S2t + (size_t)b * LQ * LC;
    const float*  Cb = C   + (size_t)b * LC * DDIM;

    v8f acc[2][4];
    const v8f z = {0.f,0.f,0.f,0.f,0.f,0.f,0.f,0.f};
#pragma unroll
    for (int i = 0; i < 2; ++i)
#pragma unroll
        for (int j = 0; j < 4; ++j) acc[i][j] = z;

    const int r  = t >> 3;
    const int ks = (t & 7) * 4;

    for (int k0 = 0; k0 < LC; k0 += KB) {
        __syncthreads();
        // A tile (f16 already): [128 rows of LQ][32 k]
#pragma unroll
        for (int i = 0; i < 4; ++i) {
            int rr = r + 32 * i;
            v4h v = *(const v4h*)(Ab + (size_t)(m0 + rr) * LC + k0 + ks);
            *(v4h*)(aT + rr * LDT + ks) = v;
        }
        // B tile: C f32 [k][n] -> LDS [n][k] (f16)
#pragma unroll
        for (int it = 0; it < 4; ++it) {
            int id = t + 256 * it;          // 0..1023
            int k  = id >> 5;               // 0..31
            int n4 = (id & 31) * 4;         // 0..124
            float4 v = *(const float4*)(Cb + (size_t)(k0 + k) * DDIM + n0 + n4);
            bT[(n4 + 0) * LDT + k] = (half_t)v.x;
            bT[(n4 + 1) * LDT + k] = (half_t)v.y;
            bT[(n4 + 2) * LDT + k] = (half_t)v.z;
            bT[(n4 + 3) * LDT + k] = (half_t)v.w;
        }
        __syncthreads();
        v16h af0 = frag_a(aT + (wm * 32 + 0)  * LDT, lane);
        v16h af1 = frag_a(aT + (wm * 32 + 16) * LDT, lane);
#pragma unroll
        for (int jn = 0; jn < 4; ++jn) {
            v16h bf = frag_b(bT + (wn * 64 + jn * 16) * LDT, lane);
            acc[0][jn] = wmma16(af0, bf, acc[0][jn]);
            acc[1][jn] = wmma16(af1, bf, acc[1][jn]);
        }
    }

    const int n_l = lane & 15;
    const int m_l = (lane >> 4) * 8;
    half_t* Tb = T16 + (size_t)b * LQ * DDIM;
#pragma unroll
    for (int im = 0; im < 2; ++im)
#pragma unroll
        for (int jn = 0; jn < 4; ++jn) {
            int nn = n0 + wn * 64 + jn * 16 + n_l;
#pragma unroll
            for (int v = 0; v < 8; ++v) {
                int mm = m0 + wm * 32 + im * 16 + m_l + v;
                Tb[(size_t)mm * DDIM + nn] = (half_t)acc[im][jn][v];
            }
        }
}

// ---------------------------------------------------------------------------
// K4: A = S1 @ Q, Bt = S1 @ T (shared A-fragments); write [C|A|C*A|C*Bt]
// grid (D/64, LC/128, B), block 256, wave tile 32x32 per GEMM
// ---------------------------------------------------------------------------
__global__ __launch_bounds__(256)
void k_gemm_out(const half_t* __restrict__ S1, const float* __restrict__ Q,
                const half_t* __restrict__ T16, const float* __restrict__ C,
                float* __restrict__ out) {
    __shared__ half_t aT[128 * LDT];
    __shared__ half_t qT[64 * LDT];
    __shared__ half_t tT[64 * LDT];
    const int b  = blockIdx.z;
    const int m0 = blockIdx.y * 128;   // over LC
    const int n0 = blockIdx.x * 64;    // over D
    const int t = threadIdx.x;
    const int lane = t & 31;
    const int wave = t >> 5;
    const int wm = wave >> 1, wn = wave & 1;

    const half_t* Ab = S1  + (size_t)b * LC * LQ;
    const float*  Qb = Q   + (size_t)b * LQ * DDIM;
    const half_t* Tb = T16 + (size_t)b * LQ * DDIM;

    v8f acc1[2][2], acc2[2][2];
    const v8f z = {0.f,0.f,0.f,0.f,0.f,0.f,0.f,0.f};
#pragma unroll
    for (int i = 0; i < 2; ++i)
#pragma unroll
        for (int j = 0; j < 2; ++j) { acc1[i][j] = z; acc2[i][j] = z; }

    const int r  = t >> 3;
    const int ks = (t & 7) * 4;

    for (int k0 = 0; k0 < LQ; k0 += KB) {
        __syncthreads();
        // A tile from S1 (f16): [128][32]
#pragma unroll
        for (int i = 0; i < 4; ++i) {
            int rr = r + 32 * i;
            v4h v = *(const v4h*)(Ab + (size_t)(m0 + rr) * LQ + k0 + ks);
            *(v4h*)(aT + rr * LDT + ks) = v;
        }
        // Q tile f32 [k][n] -> LDS [n][k]; T tile f16 [k][n] -> LDS [n][k]
#pragma unroll
        for (int it = 0; it < 2; ++it) {
            int id = t + 256 * it;          // 0..511
            int k  = id >> 4;               // 0..31
            int n4 = (id & 15) * 4;         // 0..60
            float4 v = *(const float4*)(Qb + (size_t)(k0 + k) * DDIM + n0 + n4);
            qT[(n4 + 0) * LDT + k] = (half_t)v.x;
            qT[(n4 + 1) * LDT + k] = (half_t)v.y;
            qT[(n4 + 2) * LDT + k] = (half_t)v.z;
            qT[(n4 + 3) * LDT + k] = (half_t)v.w;
            v4h h = *(const v4h*)(Tb + (size_t)(k0 + k) * DDIM + n0 + n4);
            tT[(n4 + 0) * LDT + k] = h[0];
            tT[(n4 + 1) * LDT + k] = h[1];
            tT[(n4 + 2) * LDT + k] = h[2];
            tT[(n4 + 3) * LDT + k] = h[3];
        }
        __syncthreads();
        v16h af0 = frag_a(aT + (wm * 32 + 0)  * LDT, lane);
        v16h af1 = frag_a(aT + (wm * 32 + 16) * LDT, lane);
#pragma unroll
        for (int jn = 0; jn < 2; ++jn) {
            v16h bq = frag_b(qT + (wn * 32 + jn * 16) * LDT, lane);
            v16h bt = frag_b(tT + (wn * 32 + jn * 16) * LDT, lane);
            acc1[0][jn] = wmma16(af0, bq, acc1[0][jn]);
            acc1[1][jn] = wmma16(af1, bq, acc1[1][jn]);
            acc2[0][jn] = wmma16(af0, bt, acc2[0][jn]);
            acc2[1][jn] = wmma16(af1, bt, acc2[1][jn]);
        }
    }

    const int n_l = lane & 15;
    const int m_l = (lane >> 4) * 8;
    const float* Cb = C + (size_t)b * LC * DDIM;
    float* outb = out + (size_t)b * LC * (4 * DDIM);
#pragma unroll
    for (int im = 0; im < 2; ++im)
#pragma unroll
        for (int jn = 0; jn < 2; ++jn) {
            int nn = n0 + wn * 32 + jn * 16 + n_l;
#pragma unroll
            for (int v = 0; v < 8; ++v) {
                int mm = m0 + wm * 32 + im * 16 + m_l + v;
                float a  = acc1[im][jn][v];
                float bt = acc2[im][jn][v];
                float cv = Cb[(size_t)mm * DDIM + nn];
                float* orow = outb + (size_t)mm * (4 * DDIM);
                orow[nn]             = cv;
                orow[DDIM + nn]      = a;
                orow[2 * DDIM + nn]  = cv * a;
                orow[3 * DDIM + nn]  = cv * bt;
            }
        }
}

// ---------------------------------------------------------------------------
extern "C" void kernel_launch(void* const* d_in, const int* in_sizes, int n_in,
                              void* d_out, int out_size, void* d_ws, size_t ws_size,
                              hipStream_t stream) {
    const float* C     = (const float*)d_in[0];
    const float* Q     = (const float*)d_in[1];
    const int*   Cmask = (const int*)d_in[2];
    const int*   Qmask = (const int*)d_in[3];
    const float* w4C   = (const float*)d_in[4];
    const float* w4Q   = (const float*)d_in[5];
    const float* w4mlu = (const float*)d_in[6];
    const float* bias  = (const float*)d_in[7];
    float* out = (float*)d_out;

    char* ws = (char*)d_ws;
    float*  cw  = (float*)(ws);                                  //   64 KB
    float*  qw  = (float*)(ws + (65536));                        //   16 KB
    float*  S   = (float*)(ws + (65536 + 16384));                //   16 MB
    half_t* S1  = (half_t*)(ws + (81920 + 16777216));            //    8 MB
    half_t* S2t = (half_t*)(ws + (81920 + 16777216 + 8388608));  //    8 MB
    half_t* T16 = (half_t*)(ws + (81920 + 16777216 + 16777216)); //    4 MB
    (void)in_sizes; (void)n_in; (void)out_size; (void)ws_size;

    // K0: row-scalar dots (bias folded into cw)
    k_rowdots<<<B_DIM * LC + B_DIM * LQ, 256, 0, stream>>>(C, Q, w4C, w4Q, bias, cw, qw);
    // K1: trilinear similarity S (WMMA NT GEMM + fused adds)
    k_gemm_S<<<dim3(LQ / 128, LC / 128, B_DIM), 256, 0, stream>>>(C, Q, w4mlu, cw, qw, S);
    // K2: masked softmaxes
    k_softmax_rows<<<B_DIM * LC, 256, 0, stream>>>(S, Qmask, S1);
    k_softmax_cols<<<dim3(LQ / 16, B_DIM), dim3(16, 16), 0, stream>>>(S, Cmask, S2t);
    // K3: T = S2^T @ C  (associativity: (S1 S2^T) C == S1 (S2^T C), 3x fewer FLOPs)
    k_gemm_T<<<dim3(DDIM / 128, LQ / 128, B_DIM), 256, 0, stream>>>(S2t, C, T16);
    // K4: A = S1@Q, Bt = S1@T, write [C | A | C*A | C*Bt]
    k_gemm_out<<<dim3(DDIM / 64, LC / 128, B_DIM), 256, 0, stream>>>(S1, Q, T16, C, out);
}
